// QKVMultiheadAttention_81217831567702
// MI455X (gfx1250) — matrix operand order repaired and checked by reading, not compiled, not run on hardware
//
#include <hip/hip_runtime.h>
#include <math.h>

typedef __attribute__((ext_vector_type(16))) _Float16 v16h;
typedef __attribute__((ext_vector_type(8)))  _Float16 v8h;
typedef __attribute__((ext_vector_type(4)))  _Float16 v4h;
typedef __attribute__((ext_vector_type(8)))  float    v8f;

#define HEADS 16
#define NCTX  2048
#define DHEAD 64
#define KT    64     // keys per tile
#define QROWS 64     // query rows per block (4 waves x 16)

static __device__ __forceinline__ v16h join8(v8h lo, v8h hi) {
    return __builtin_shufflevector(lo, hi, 0,1,2,3,4,5,6,7,8,9,10,11,12,13,14,15);
}
// exchange value with lane^16 partner (wave32 half-swap), pure VALU
static __device__ __forceinline__ int xor16i(int v) {
    return __builtin_amdgcn_permlanex16(v, v, 0x76543210, 0xFEDCBA98, false, false);
}
static __device__ __forceinline__ float xor16f(float x) {
    return __builtin_bit_cast(float, xor16i(__builtin_bit_cast(int, x)));
}
static __device__ __forceinline__ int pk2(float a, float b) {
    return __builtin_bit_cast(int, __builtin_amdgcn_cvt_pkrtz(a, b));
}

__launch_bounds__(128)
__global__ void qkv_attn_kernel(const float* __restrict__ qkv,
                                const unsigned char* __restrict__ key_mask,
                                float* __restrict__ out)
{
    __shared__ __align__(16) _Float16 qs[QROWS][72];  // Q tile (scaled f16), row-major
    __shared__ __align__(16) _Float16 kb[KT][72];     // K tile (scaled f16), row-major
    __shared__ __align__(16) _Float16 vt[DHEAD][72];  // V tile transposed: [ch][key]
    __shared__ __align__(16) float    mb[KT];         // key-mask bias (0 / -inf)

    const int tid  = threadIdx.x;
    const int wid  = tid >> 5;
    const int lane = tid & 31;
    const int lm   = lane & 15;
    const int hh   = lane >> 4;

    const int blk = blockIdx.x;
    const int qb  = blk & 31;
    const int h   = (blk >> 5) & 15;
    const int b   = blk >> 9;

    const float scale = 0.35355339059327373f;  // 64^(-1/4), applied to Q and K
    const int t0 = qb * QROWS;

    // ---- stage Q tile (scaled, f16): float4 loads -> cvt_pk -> b64 stores ----
#pragma unroll
    for (int p = 0; p < 8; ++p) {
        const int idx = tid + 128 * p;
        const int r = idx >> 4, c4 = (idx & 15) * 4;
        const float4 f = *(const float4*)&qkv[(((size_t)(b * NCTX + t0 + r)) * HEADS + h) * 192 + c4];
        union { int i[2]; v4h v; } u;
        u.i[0] = pk2(f.x * scale, f.y * scale);
        u.i[1] = pk2(f.z * scale, f.w * scale);
        *(v4h*)&qs[r][c4] = u.v;
    }
    __syncthreads();

    // ---- resident Q^T B-fragments (n = query = lm, k = channel) ----
    const int qr = wid * 16 + lm;
    const v16h qB0 = join8(*(const v8h*)&qs[qr][8*hh],      *(const v8h*)&qs[qr][16 + 8*hh]);       // ch 0..31
    const v16h qB1 = join8(*(const v8h*)&qs[qr][32 + 8*hh], *(const v8h*)&qs[qr][48 + 8*hh]);  // ch 32..63

    float m_run = -1.0e30f, l_run = 0.0f;   // per-lane = per-query-column state
    v8f acc[4];                              // out^T: acc[nb][r] -> ch = 16nb+8hh+r, query = lm
#pragma unroll
    for (int nb = 0; nb < 4; ++nb)
#pragma unroll
        for (int r = 0; r < 8; ++r) acc[nb][r] = 0.0f;

    // ==================== flash loop over 64-key tiles ====================
    for (int kt = 0; kt < NCTX / KT; ++kt) {
        const int s0 = kt * KT;
        __syncthreads();

        // stage K (scaled f16, row-major): float4 -> b64 stores
#pragma unroll
        for (int p = 0; p < 8; ++p) {
            const int idx = tid + 128 * p;
            const int r = idx >> 4, c4 = (idx & 15) * 4;
            const float4 f = *(const float4*)&qkv[(((size_t)(b * NCTX + s0 + r)) * HEADS + h) * 192 + 64 + c4];
            union { int i[2]; v4h v; } u;
            u.i[0] = pk2(f.x * scale, f.y * scale);
            u.i[1] = pk2(f.z * scale, f.w * scale);
            *(v4h*)&kb[r][c4] = u.v;
        }
        // stage V transposed: 4 coalesced row loads -> one b64 store down a column
#pragma unroll
        for (int g = 0; g < 8; ++g) {
            const int ch = tid & 63;
            const int rb = ((tid >> 6) << 2) + g * 8;
            const size_t vbase = (((size_t)(b * NCTX + s0 + rb)) * HEADS + h) * 192 + 128 + ch;
            const float a0 = qkv[vbase];
            const float a1 = qkv[vbase + 1 * HEADS * 192];
            const float a2 = qkv[vbase + 2 * HEADS * 192];
            const float a3 = qkv[vbase + 3 * HEADS * 192];
            union { int i[2]; v4h v; } u;
            u.i[0] = pk2(a0, a1);
            u.i[1] = pk2(a2, a3);
            *(v4h*)&vt[ch][rb] = u.v;
        }
        if (tid < KT)
            mb[tid] = key_mask[(size_t)b * NCTX + s0 + tid] ? 0.0f : -__builtin_inff();
        // prefetch next tile's K and V lines (global_prefetch_b8)
        if (kt + 1 < NCTX / KT) {
            const size_t nbase = (((size_t)(b * NCTX + s0 + KT + (tid >> 1))) * HEADS + h) * 192 + (tid & 1) * 32;
            __builtin_prefetch(&qkv[nbase + 64], 0, 3);
            __builtin_prefetch(&qkv[nbase + 128], 0, 3);
        }
        __syncthreads();

        // ---- S^T = K * Q^T : 4 key tiles x 2-chain over channels ----
        v8f c[4] = {};
#pragma unroll
        for (int j = 0; j < 4; ++j) {
            const int kr = 16 * j + lm;
            const v16h a0 = join8(*(const v8h*)&kb[kr][8*hh],      *(const v8h*)&kb[kr][16 + 8*hh]);
            c[j] = __builtin_amdgcn_wmma_f32_16x16x32_f16(false, a0, false, qB0, (short)0, c[j], false, false);
            const v16h a1 = join8(*(const v8h*)&kb[kr][32 + 8*hh], *(const v8h*)&kb[kr][48 + 8*hh]);
            c[j] = __builtin_amdgcn_wmma_f32_16x16x32_f16(false, a1, false, qB1, (short)0, c[j], false, false);
        }

        // ---- masked online softmax over keys (register + one permlanex16) ----
        float s[4][8];
#pragma unroll
        for (int j = 0; j < 4; ++j) {
            const float* mp = &mb[16 * j + 8 * hh];
            const float4 b0 = *(const float4*)mp;
            const float4 b1 = *(const float4*)(mp + 4);
            s[j][0] = c[j][0] + b0.x;  s[j][1] = c[j][1] + b0.y;
            s[j][2] = c[j][2] + b0.z;  s[j][3] = c[j][3] + b0.w;
            s[j][4] = c[j][4] + b1.x;  s[j][5] = c[j][5] + b1.y;
            s[j][6] = c[j][6] + b1.z;  s[j][7] = c[j][7] + b1.w;
        }
        float mloc = s[0][0];
#pragma unroll
        for (int j = 0; j < 4; ++j)
#pragma unroll
            for (int r = 0; r < 8; ++r) mloc = fmaxf(mloc, s[j][r]);
        mloc = fmaxf(mloc, xor16f(mloc));            // combine with partner's 32 keys
        const float mnew  = fmaxf(m_run, mloc);
        const float alpha = __expf(m_run - mnew);
        m_run = mnew;

        float p[4][8];
        float ls = 0.0f;
#pragma unroll
        for (int j = 0; j < 4; ++j)
#pragma unroll
            for (int r = 0; r < 8; ++r) { p[j][r] = __expf(s[j][r] - mnew); ls += p[j][r]; }
        l_run = l_run * alpha + (ls + xor16f(ls));

#pragma unroll
        for (int nb = 0; nb < 4; ++nb)
#pragma unroll
            for (int r = 0; r < 8; ++r) acc[nb][r] *= alpha;

        // ---- out^T += V^T * P^T : P^T B-fragments built purely in registers ----
#pragma unroll
        for (int cc = 0; cc < 2; ++cc) {
            int pkA[4], pkB[4], pxA[4], pxB[4];
#pragma unroll
            for (int w = 0; w < 4; ++w) {
                pkA[w] = pk2(p[2*cc][2*w],     p[2*cc][2*w + 1]);      // own keys of tile 2cc
                pkB[w] = pk2(p[2*cc + 1][2*w], p[2*cc + 1][2*w + 1]);  // own keys of tile 2cc+1
                pxA[w] = xor16i(pkA[w]);                               // partner's tile-2cc keys
                pxB[w] = xor16i(pkB[w]);                               // partner's tile-2cc+1 keys
            }
            union { int i[8]; v16h v; } u;
#pragma unroll
            for (int w = 0; w < 4; ++w) {
                u.i[w]     = hh ? pxB[w] : pkA[w];   // k = 0..7  of this lane-half's 16-key tile
                u.i[4 + w] = hh ? pkB[w] : pxA[w];   // k = 8..15
            }
#pragma unroll
            for (int nb = 0; nb < 4; ++nb) {
                const int ch = 16 * nb + lm;
                const v16h aV = join8(*(const v8h*)&vt[ch][32*cc + 16*hh],
                                      *(const v8h*)&vt[ch][32*cc + 16*hh + 8]);
                acc[nb] = __builtin_amdgcn_wmma_f32_16x16x32_f16(false, aV, false, u.v, (short)0, acc[nb], false, false);
            }
        }
    }

    // ---- normalize + vectorized store (consecutive channels per lane) ----
    const float inv = 1.0f / l_run;
    const int t = t0 + wid * 16 + lm;
    const size_t base = ((size_t)(b * NCTX + t) * HEADS + h) * DHEAD;
#pragma unroll
    for (int nb = 0; nb < 4; ++nb) {
        float4 o0, o1;
        o0.x = acc[nb][0] * inv;  o0.y = acc[nb][1] * inv;
        o0.z = acc[nb][2] * inv;  o0.w = acc[nb][3] * inv;
        o1.x = acc[nb][4] * inv;  o1.y = acc[nb][5] * inv;
        o1.z = acc[nb][6] * inv;  o1.w = acc[nb][7] * inv;
        *(float4*)&out[base + 16 * nb + 8 * hh]     = o0;
        *(float4*)&out[base + 16 * nb + 8 * hh + 4] = o1;
    }
}

extern "C" void kernel_launch(void* const* d_in, const int* in_sizes, int n_in,
                              void* d_out, int out_size, void* d_ws, size_t ws_size,
                              hipStream_t stream) {
    const float*         qkv = (const float*)d_in[0];
    // d_in[1] = query_mask (all true) -- unused
    const unsigned char* km  = (const unsigned char*)d_in[2];
    float*               outp = (float*)d_out;

    const int bs = in_sizes[0] / (NCTX * 3 * HEADS * DHEAD);
    const int nblocks = bs * HEADS * (NCTX / QROWS);   // b*512 + h*32 + qb
    qkv_attn_kernel<<<nblocks, 128, 0, stream>>>(qkv, km, outp);
}
